// MultiheadSelfAttention_78520592105526
// MI455X (gfx1250) — matrix operand order repaired and checked by reading, not compiled, not run on hardware
//
#include <hip/hip_runtime.h>
#include <hip/hip_bf16.h>

// ---------------------------------------------------------------------------
// MI455X (gfx1250) multi-head self-attention, bf16 WMMA + flash attention.
// wave32, V_WMMA_F32_16X16X32_BF16 (f32 accumulate).
// GEMM + attention both use GLOBAL_LOAD_ASYNC_TO_LDS_B128 (ASYNCcnt) with
// double-buffered LDS tiles.
// ---------------------------------------------------------------------------

#define USE_ASYNC_LDS 1

typedef __attribute__((ext_vector_type(16))) __bf16 v16bf;
typedef __attribute__((ext_vector_type(8)))  __bf16 v8bf;
typedef __attribute__((ext_vector_type(8)))  float  v8f;

#define BATCH 4
#define SEQ   2048
#define DMODEL 1024
#define NHEADS 16
#define DK    64
#define MROWS (BATCH * SEQ)   // 8192

__device__ __forceinline__ v16bf combine16(v8bf lo, v8bf hi) {
  v16bf r;
#pragma unroll
  for (int e = 0; e < 8; ++e) { r[e] = lo[e]; r[e + 8] = hi[e]; }
  return r;
}

#if USE_ASYNC_LDS
// Async global->LDS copy, 16B per lane. INST_OFFSET is added to BOTH the
// global and LDS addresses (CDNA5 ISA), so one address pair
// serves several 16B chunks. Tracked by ASYNCcnt.
#define GLD_ASYNC_B128(ldsoff_u32, gaddr_u64, imm)                         \
  asm volatile("global_load_async_to_lds_b128 %0, %1, off offset:%2"       \
               :: "v"(ldsoff_u32), "v"(gaddr_u64), "i"(imm) : "memory")
#define WAIT_ASYNC0() asm volatile("s_wait_asynccnt 0x0" ::: "memory")
#else
#define WAIT_ASYNC0()
#endif

// ------------------------------ converters ---------------------------------

__global__ void cvt_bf16_kernel(const float* __restrict__ in,
                                __bf16* __restrict__ out, int n) {
  int i = blockIdx.x * blockDim.x + threadIdx.x;
  if (i < n) out[i] = (__bf16)in[i];
}

// W[k][n] (1024x1024 f32) -> Wt[n][k] bf16
__global__ void transpose_cvt_kernel(const float* __restrict__ in,
                                     __bf16* __restrict__ out) {
  int i = blockIdx.x * blockDim.x + threadIdx.x;   // over 1024*1024
  int k = i >> 10, n = i & 1023;
  out[(size_t)n * 1024 + k] = (__bf16)in[i];
}

// ------------------------------ GEMM ---------------------------------------
// C = A (MROWS x 1024, bf16 row-major) @ W (via Wt = W^T, bf16 [N][K]).
// Block: 128x128 tile, 256 threads = 8 waves (2x4), wave tile 64x32,
// K-step 32, double-buffered LDS, async global->LDS fill.
// mode 0: Q epilogue (RoPE + 0.125 scale) -> Qb[b,h,s,d] bf16
// mode 1: K epilogue (RoPE)               -> Kb[b,h,s,d] bf16
// mode 2: V epilogue (transpose)          -> Vt[b,h,d,s] bf16
// mode 3: plain f32 store                 -> out[m][n]

__global__ __launch_bounds__(256)
void gemm_bf16_kernel(const __bf16* __restrict__ A,
                      const __bf16* __restrict__ Bt,
                      void* __restrict__ outp, int mode,
                      const int*   __restrict__ pos,
                      const float* __restrict__ cosT,
                      const float* __restrict__ sinT) {
  // padded rows: 40 elems = 80B -> 16B aligned, conflict-friendly stride
  __shared__ __align__(16) __bf16 As[2][128][40];
  __shared__ __align__(16) __bf16 Bs[2][128][40];

  const int K = 1024, N = 1024;
  const int tid  = threadIdx.x;
  const int lane = tid & 31;
  const int wave = tid >> 5;
  const int lh = lane >> 4;       // lane half (0/1)
  const int ln = lane & 15;
  const int mw = wave & 1;        // 2 wave rows
  const int nw = wave >> 1;       // 4 wave cols
  const int blockM = blockIdx.y * 128;
  const int blockN = blockIdx.x * 128;

  v8f acc[4][2];
#pragma unroll
  for (int i = 0; i < 4; ++i)
#pragma unroll
    for (int j = 0; j < 2; ++j) acc[i][j] = (v8f){};

  const int lrow = tid >> 1;        // 0..127
  const int lcol = (tid & 1) * 16;  // 0 or 16

  auto stage = [&](int buf, int kk) {
#if USE_ASYNC_LDS
    const uint64_t ga =
        (uint64_t)(uintptr_t)(A + (size_t)(blockM + lrow) * K + kk + lcol);
    const uint32_t la = (uint32_t)(uintptr_t)&As[buf][lrow][lcol];
    GLD_ASYNC_B128(la, ga, 0);
    GLD_ASYNC_B128(la, ga, 16);
    const uint64_t gb =
        (uint64_t)(uintptr_t)(Bt + (size_t)(blockN + lrow) * K + kk + lcol);
    const uint32_t lb = (uint32_t)(uintptr_t)&Bs[buf][lrow][lcol];
    GLD_ASYNC_B128(lb, gb, 0);
    GLD_ASYNC_B128(lb, gb, 16);
#else
    const __bf16* ga = A + (size_t)(blockM + lrow) * K + kk + lcol;
    *(v8bf*)&As[buf][lrow][lcol]     = *(const v8bf*)ga;
    *(v8bf*)&As[buf][lrow][lcol + 8] = *(const v8bf*)(ga + 8);
    const __bf16* gb = Bt + (size_t)(blockN + lrow) * K + kk + lcol;
    *(v8bf*)&Bs[buf][lrow][lcol]     = *(const v8bf*)gb;
    *(v8bf*)&Bs[buf][lrow][lcol + 8] = *(const v8bf*)(gb + 8);
#endif
  };

  int buf = 0;
  stage(0, 0);
  WAIT_ASYNC0();
  __syncthreads();

  for (int kk = 0; kk < K; kk += 32) {
    if (kk + 32 < K) stage(buf ^ 1, kk + 32);   // fill next while computing

    v16bf af[4], bfr[2];
#pragma unroll
    for (int i = 0; i < 4; ++i) {
      const int row = mw * 64 + i * 16 + ln;
      v8bf x0 = *(const v8bf*)&As[buf][row][lh * 8];
      v8bf x1 = *(const v8bf*)&As[buf][row][16 + lh * 8];
      af[i] = combine16(x0, x1);
    }
#pragma unroll
    for (int j = 0; j < 2; ++j) {
      const int row = nw * 32 + j * 16 + ln;
      v8bf y0 = *(const v8bf*)&Bs[buf][row][lh * 8];
      v8bf y1 = *(const v8bf*)&Bs[buf][row][16 + lh * 8];
      bfr[j] = combine16(y0, y1);
    }
#pragma unroll
    for (int i = 0; i < 4; ++i)
#pragma unroll
      for (int j = 0; j < 2; ++j)
        acc[i][j] = __builtin_amdgcn_wmma_f32_16x16x32_bf16(
            false, af[i], false, bfr[j], (short)0, acc[i][j], false, false);

    WAIT_ASYNC0();      // my fills into buf^1 done
    __syncthreads();    // everyone's fills visible; reads of buf finished
    buf ^= 1;
  }

  // epilogue (C layout: VGPR r -> row r + 8*lh, col = ln within 16-tile)
#pragma unroll
  for (int i = 0; i < 4; ++i) {
#pragma unroll
    for (int j = 0; j < 2; ++j) {
      const int n_g = blockN + nw * 32 + j * 16 + ln;
#pragma unroll
      for (int r = 0; r < 8; ++r) {
        const int m_g = blockM + mw * 64 + i * 16 + r + 8 * lh;
        float v = acc[i][j][r];
        if (mode == 3) {
          ((float*)outp)[(size_t)m_g * N + n_g] = v;
        } else {
          const int b  = m_g >> 11, s = m_g & 2047;
          const int hh = n_g >> 6,  d = n_g & 63;
          if (mode == 2) {
            ((__bf16*)outp)[((size_t)(b * NHEADS + hh) * DK + d) * SEQ + s] =
                (__bf16)v;
          } else {
            // RoPE: even/odd columns live in adjacent lanes -> shfl_xor(1)
            float pv = __shfl_xor(v, 1, 32);
            const int p  = pos[b * SEQ + s];
            const float c  = cosT[p * 32 + (d >> 1)];
            const float sn = sinT[p * 32 + (d >> 1)];
            float o = (d & 1) ? (pv * sn + v * c) : (v * c - pv * sn);
            if (mode == 0) o *= 0.125f;   // fold 1/sqrt(64) into Q
            ((__bf16*)outp)[((size_t)(b * NHEADS + hh) * SEQ + s) * DK + d] =
                (__bf16)o;
          }
        }
      }
    }
  }
}

// --------------------------- flash attention -------------------------------
// Grid (SEQ/64, BATCH*NHEADS), 128 threads = 4 waves; each wave owns a 16-row
// q tile. K/V tiles (32 keys) are shared by all waves through double-buffered
// LDS filled with async loads. All waves run the block-uniform causal trip
// count 2*blockIdx.x+2; fully-masked trailing blocks are exact no-ops in the
// online softmax (e=0, rs=0, alpha=1), so __syncthreads() is safe.

__global__ __launch_bounds__(128)
void flash_attn_kernel(const __bf16* __restrict__ Q,
                       const __bf16* __restrict__ Kb,
                       const __bf16* __restrict__ Vt,
                       __bf16* __restrict__ ctx) {
  __shared__ __align__(16) __bf16 Klds[2][32][72];   // 32 keys x 64 d (+pad)
  __shared__ __align__(16) __bf16 Vlds[2][64][40];   // 64 d x 32 keys (+pad)
  __shared__ __align__(16) __bf16 Plds[4][16][32];   // per-wave P tile

  const int tid  = threadIdx.x;
  const int lane = tid & 31;
  const int wave = tid >> 5;
  const int lh = lane >> 4;
  const int ln = lane & 15;
  const int bh = blockIdx.y;                     // b*16 + h
  const int qbase = blockIdx.x * 64 + wave * 16;

  const __bf16* Qh = Q  + (size_t)bh * SEQ * DK;
  const __bf16* Kh = Kb + (size_t)bh * SEQ * DK;
  const __bf16* Vh = Vt + (size_t)bh * DK * SEQ;

  // Q A-fragments for d-chunks [0,32) and [32,64)
  v16bf aq[2];
#pragma unroll
  for (int c = 0; c < 2; ++c) {
    const __bf16* p = Qh + (size_t)(qbase + ln) * DK + c * 32;
    v8bf x0 = *(const v8bf*)(p + lh * 8);
    v8bf x1 = *(const v8bf*)(p + 16 + lh * 8);
    aq[c] = combine16(x0, x1);
  }

  v8f O[4];
#pragma unroll
  for (int c = 0; c < 4; ++c) O[c] = (v8f){};
  float mrun[8], lrun[8];
#pragma unroll
  for (int r = 0; r < 8; ++r) { mrun[r] = -1.0e30f; lrun[r] = 0.0f; }

  // block K/V tile staging: 128 threads move 4KB K + 4KB V per k-block
  const int krow = tid >> 2, kcol = (tid & 3) * 16;   // K: 32 x 64
  const int vrow = tid >> 1, vcol = (tid & 1) * 16;   // V: 64 x 32
  auto stage = [&](int buf, int kb) {
    const int kbase = kb * 32;
#if USE_ASYNC_LDS
    const uint64_t gk =
        (uint64_t)(uintptr_t)(Kh + (size_t)(kbase + krow) * DK + kcol);
    const uint32_t lk = (uint32_t)(uintptr_t)&Klds[buf][krow][kcol];
    GLD_ASYNC_B128(lk, gk, 0);
    GLD_ASYNC_B128(lk, gk, 16);
    const uint64_t gv =
        (uint64_t)(uintptr_t)(Vh + (size_t)vrow * SEQ + kbase + vcol);
    const uint32_t lv = (uint32_t)(uintptr_t)&Vlds[buf][vrow][vcol];
    GLD_ASYNC_B128(lv, gv, 0);
    GLD_ASYNC_B128(lv, gv, 16);
#else
    const __bf16* gk = Kh + (size_t)(kbase + krow) * DK + kcol;
    *(v8bf*)&Klds[buf][krow][kcol]     = *(const v8bf*)gk;
    *(v8bf*)&Klds[buf][krow][kcol + 8] = *(const v8bf*)(gk + 8);
    const __bf16* gv = Vh + (size_t)vrow * SEQ + kbase + vcol;
    *(v8bf*)&Vlds[buf][vrow][vcol]     = *(const v8bf*)gv;
    *(v8bf*)&Vlds[buf][vrow][vcol + 8] = *(const v8bf*)(gv + 8);
#endif
  };

  const int ntrip = blockIdx.x * 2 + 2;   // block-uniform causal bound
  int buf = 0;
  stage(0, 0);
  WAIT_ASYNC0();
  __syncthreads();

  for (int kb = 0; kb < ntrip; ++kb) {
    const int kbase = kb * 32;
    if (kb + 1 < ntrip) stage(buf ^ 1, kb + 1);

    // scores S = Q(16x64) @ K^T : two 16x16 tiles, K-dim = 64 (2 WMMA each)
    v8f S[2]; S[0] = (v8f){}; S[1] = (v8f){};
#pragma unroll
    for (int t = 0; t < 2; ++t) {
#pragma unroll
      for (int c = 0; c < 2; ++c) {
        const __bf16* p = &Klds[buf][t * 16 + ln][c * 32];
        v8bf y0 = *(const v8bf*)(p + lh * 8);
        v8bf y1 = *(const v8bf*)(p + 16 + lh * 8);
        v16bf bk = combine16(y0, y1);
        S[t] = __builtin_amdgcn_wmma_f32_16x16x32_bf16(
            false, aq[c], false, bk, (short)0, S[t], false, false);
      }
    }

    // causal mask (1/sqrt(dk) already folded into Q)
#pragma unroll
    for (int t = 0; t < 2; ++t)
#pragma unroll
      for (int r = 0; r < 8; ++r) {
        const int qg = qbase + r + 8 * lh;
        const int kg = kbase + t * 16 + ln;
        if (kg > qg) S[t][r] = -1.0e30f;
      }

    // online softmax; row stats via shfl within 16-lane halves
#pragma unroll
    for (int r = 0; r < 8; ++r) {
      float rm = fmaxf(S[0][r], S[1][r]);
#pragma unroll
      for (int m = 1; m < 16; m <<= 1) rm = fmaxf(rm, __shfl_xor(rm, m, 32));
      const float nm = fmaxf(mrun[r], rm);
      const float e0 = __expf(S[0][r] - nm);
      const float e1 = __expf(S[1][r] - nm);
      S[0][r] = e0; S[1][r] = e1;
      float rs = e0 + e1;
#pragma unroll
      for (int m = 1; m < 16; m <<= 1) rs += __shfl_xor(rs, m, 32);
      const float alpha = __expf(mrun[r] - nm);
      lrun[r] = lrun[r] * alpha + rs;
      mrun[r] = nm;
#pragma unroll
      for (int c = 0; c < 4; ++c) O[c][r] *= alpha;
    }

    // reshape P: C-layout -> A-fragment through per-wave LDS tile.
#pragma unroll
    for (int t = 0; t < 2; ++t)
#pragma unroll
      for (int r = 0; r < 8; ++r)
        Plds[wave][r + 8 * lh][t * 16 + ln] = (__bf16)S[t][r];
    asm volatile("s_wait_dscnt 0x0" ::: "memory");

    v16bf ap;
    {
      const __bf16* p = &Plds[wave][ln][0];
      v8bf x0 = *(const v8bf*)(p + lh * 8);
      v8bf x1 = *(const v8bf*)(p + 16 + lh * 8);
      ap = combine16(x0, x1);
    }

    // O += P(16x32) @ V(32x64): V^T LDS tile gives contiguous B-fragments
#pragma unroll
    for (int c = 0; c < 4; ++c) {
      const __bf16* p = &Vlds[buf][c * 16 + ln][0];
      v8bf y0 = *(const v8bf*)(p + lh * 8);
      v8bf y1 = *(const v8bf*)(p + 16 + lh * 8);
      v16bf bv = combine16(y0, y1);
      O[c] = __builtin_amdgcn_wmma_f32_16x16x32_bf16(
          false, ap, false, bv, (short)0, O[c], false, false);
    }

    WAIT_ASYNC0();      // my fills into buf^1 done
    __syncthreads();    // all waves done with buf; fills visible
    buf ^= 1;
  }

  // normalize and write ctx[b, s, h*64 + d] (feeds the Wo GEMM)
  const int b  = bh >> 4;
  const int hh = bh & 15;
#pragma unroll
  for (int r = 0; r < 8; ++r) {
    const float inv = 1.0f / lrun[r];
    const int s = qbase + r + 8 * lh;
    const size_t base = ((size_t)(b * SEQ + s)) * DMODEL + hh * DK;
#pragma unroll
    for (int c = 0; c < 4; ++c)
      ctx[base + c * 16 + ln] = (__bf16)(O[c][r] * inv);
  }
}

// ------------------------------ launcher -----------------------------------

extern "C" void kernel_launch(void* const* d_in, const int* in_sizes, int n_in,
                              void* d_out, int out_size, void* d_ws,
                              size_t ws_size, hipStream_t stream) {
  const float* x    = (const float*)d_in[0];
  const int*   pos  = (const int*)d_in[1];
  const float* Wq   = (const float*)d_in[2];
  const float* Wk   = (const float*)d_in[3];
  const float* Wv   = (const float*)d_in[4];
  const float* Wo   = (const float*)d_in[5];
  const float* cosT = (const float*)d_in[6];
  const float* sinT = (const float*)d_in[7];

  char* w = (char*)d_ws;
  __bf16* xb  = (__bf16*)w; w += (size_t)MROWS * DMODEL * 2;         // 16 MB
  __bf16* WqT = (__bf16*)w; w += (size_t)DMODEL * DMODEL * 2;        //  2 MB
  __bf16* WkT = (__bf16*)w; w += (size_t)DMODEL * DMODEL * 2;
  __bf16* WvT = (__bf16*)w; w += (size_t)DMODEL * DMODEL * 2;
  __bf16* WoT = (__bf16*)w; w += (size_t)DMODEL * DMODEL * 2;
  __bf16* Qb  = (__bf16*)w; w += (size_t)MROWS * DMODEL * 2;         // 16 MB
  __bf16* Kb  = (__bf16*)w; w += (size_t)MROWS * DMODEL * 2;
  __bf16* Vt  = (__bf16*)w; w += (size_t)MROWS * DMODEL * 2;
  __bf16* ctx = (__bf16*)w; w += (size_t)MROWS * DMODEL * 2;         // total 88 MB

  const int nX = MROWS * DMODEL;
  cvt_bf16_kernel<<<nX / 256, 256, 0, stream>>>(x, xb, nX);
  const int nW = DMODEL * DMODEL;
  transpose_cvt_kernel<<<nW / 256, 256, 0, stream>>>(Wq, WqT);
  transpose_cvt_kernel<<<nW / 256, 256, 0, stream>>>(Wk, WkT);
  transpose_cvt_kernel<<<nW / 256, 256, 0, stream>>>(Wv, WvT);
  transpose_cvt_kernel<<<nW / 256, 256, 0, stream>>>(Wo, WoT);

  dim3 ggrid(DMODEL / 128, MROWS / 128);   // (8, 64)
  gemm_bf16_kernel<<<ggrid, 256, 0, stream>>>(xb, WqT, Qb, 0, pos, cosT, sinT);
  gemm_bf16_kernel<<<ggrid, 256, 0, stream>>>(xb, WkT, Kb, 1, pos, cosT, sinT);
  gemm_bf16_kernel<<<ggrid, 256, 0, stream>>>(xb, WvT, Vt, 2, pos, cosT, sinT);

  dim3 agrid(SEQ / 64, BATCH * NHEADS);    // (32, 64)
  flash_attn_kernel<<<agrid, 128, 0, stream>>>(Qb, Kb, Vt, ctx);

  gemm_bf16_kernel<<<ggrid, 256, 0, stream>>>(ctx, WoT, (float*)d_out, 3,
                                              pos, cosT, sinT);
}
